// ZyboGPT_83803401879883
// MI455X (gfx1250) — 8-bit matrix operand order repaired and checked by reading, compiled, not run on hardware
//
#include <hip/hip_runtime.h>
#include <cstdint>
#include <cstddef>


typedef int v8i __attribute__((ext_vector_type(8)));
typedef int v4i __attribute__((ext_vector_type(4)));

#define B_ 4
#define T_ 1024
#define DM 64
#define NH 2
#define HD 32
#define DFF 256
#define NL 4
#define VOC 8192
#define NTOK (B_ * T_)

__device__ __forceinline__ int imin(int a, int b) { return a < b ? a : b; }
__device__ __forceinline__ int imax(int a, int b) { return a > b ? a : b; }
__device__ __forceinline__ int clamp8(int v) { return imin(imax(v, -128), 127); }
__device__ __forceinline__ int8_t wrap8(int v) { return (int8_t)(uint8_t)(unsigned)v; }

__global__ __launch_bounds__(256) void k_quant_ternary(const float* __restrict__ w,
                                                       int8_t* __restrict__ o, int n) {
  __shared__ float red[256];
  const float* wm = w + (size_t)blockIdx.x * n;
  int8_t* om = o + (size_t)blockIdx.x * n;
  float s = 0.f;
  for (int i = threadIdx.x; i < n; i += 256) s += fabsf(wm[i]);
  red[threadIdx.x] = s;
  __syncthreads();
  for (int st = 128; st > 0; st >>= 1) {
    if ((int)threadIdx.x < st) red[threadIdx.x] += red[threadIdx.x + st];
    __syncthreads();
  }
  float scale = fmaxf(red[0] / (float)n, 1e-5f);
  for (int i = threadIdx.x; i < n; i += 256) {
    float t = rintf(wm[i] / scale);
    t = fminf(fmaxf(t, -1.f), 1.f);
    om[i] = (int8_t)(int)t;
  }
}

__global__ __launch_bounds__(256) void k_quant_emb(const float* __restrict__ e,
                                                   int8_t* __restrict__ hi,
                                                   int8_t* __restrict__ lo, int n) {
  int i = blockIdx.x * 256 + threadIdx.x;
  if (i >= n) return;
  int v = (int)rintf(e[i] * 1024.f);
  v = imin(imax(v, -32768), 32767);
  hi[i] = (int8_t)(v >> 8);
  lo[i] = (int8_t)(v & 255);
}

__global__ __launch_bounds__(256) void k_embed(const int* __restrict__ tokens,
                                               const float* __restrict__ te,
                                               const float* __restrict__ pe,
                                               int8_t* __restrict__ x) {
  int t = blockIdx.x * 256 + threadIdx.x;
  int tt = t & (T_ - 1);
  int tok = tokens[t];
  for (int c = 0; c < DM; ++c) {
    int tq = (int)rintf(te[(size_t)tok * DM + c] * 1024.f);
    tq = imin(imax(tq, -32768), 32767);
    int pq = (int)rintf(pe[(size_t)tt * DM + c] * 1024.f);
    pq = imin(imax(pq, -32768), 32767);
    x[(size_t)t * DM + c] = wrap8((tq + pq) >> 3);
  }
}

__global__ __launch_bounds__(256) void k_rmsnorm(const int8_t* __restrict__ x,
                                                 const float* __restrict__ w,
                                                 int8_t* __restrict__ out) {
  int t = blockIdx.x * 256 + threadIdx.x;
  const int8_t* xr = x + (size_t)t * DM;
  int ss = 0;
  for (int c = 0; c < DM; ++c) { int xv = xr[c]; ss += xv * xv; }
  int mean_sq = ss >> 6;
  int idx = imin(mean_sq >> 6, 255);
  float bc = (float)(idx * 64 + 32);
  int inv_rms = (int)rintf(16384.f / sqrtf(bc));
  inv_rms = imin(imax(inv_rms, 0), 16383);
  for (int c = 0; c < DM; ++c) {
    float wc = fminf(fmaxf(w[c], -2.f), 2.f);
    int gq = imin(imax((int)rintf(wc * 1024.f), -32768), 32767);
    int p1 = ((int)xr[c] * inv_rms) >> 8;
    int y = (p1 * gq) >> 10;
    out[(size_t)t * DM + c] = (int8_t)clamp8(y);
  }
}

__global__ __launch_bounds__(256) void k_vtrans(const int8_t* __restrict__ v,
                                                int8_t* __restrict__ vt) {
  int t = blockIdx.x * 256 + threadIdx.x;
  int b = t >> 10, tt = t & (T_ - 1);
  for (int c = 0; c < DM; ++c) {
    int hh = c >> 5, d = c & 31;
    vt[((size_t)((b * NH + hh) * HD + d)) * T_ + tt] = v[(size_t)t * DM + c];
  }
}

__global__ __launch_bounds__(32) void k_bitlinear(const int8_t* __restrict__ A,
                                                  const int8_t* __restrict__ W,
                                                  int8_t* __restrict__ out,
                                                  int8_t* __restrict__ resid,
                                                  int K, int mode) {
  const int m0 = blockIdx.x * 16, n0 = blockIdx.y * 16;
  const int lane = threadIdx.x, lo = lane & 15, hi = lane >> 4;
  const int N = gridDim.y * 16;
  v8i acc = {};
  for (int kc = 0; kc < K; kc += 64) {
    v8i a, b;
    const int8_t* pa = A + (size_t)(m0 + lo) * K + kc + hi * 16;
    a[0] = *(const int*)(pa + 0);  a[1] = *(const int*)(pa + 4);
    a[2] = *(const int*)(pa + 8);  a[3] = *(const int*)(pa + 12);
    a[4] = *(const int*)(pa + 32); a[5] = *(const int*)(pa + 36);
    a[6] = *(const int*)(pa + 40); a[7] = *(const int*)(pa + 44);
    const int8_t* pb = W + (size_t)(n0 + lo) * K + kc + hi * 16;
    b[0] = *(const int*)(pb + 0);  b[1] = *(const int*)(pb + 4);
    b[2] = *(const int*)(pb + 8);  b[3] = *(const int*)(pb + 12);
    b[4] = *(const int*)(pb + 32); b[5] = *(const int*)(pb + 36);
    b[6] = *(const int*)(pb + 40); b[7] = *(const int*)(pb + 44);
    acc = __builtin_amdgcn_wmma_i32_16x16x64_iu8(true, a, true, b, acc, false, false);
  }
#pragma unroll
  for (int v = 0; v < 8; ++v) {
    int r = v + hi * 8;
    int val = clamp8(acc[v] >> 6);
    if (mode == 1) val = imax(val, 0);
    size_t idx = (size_t)(m0 + r) * N + n0 + lo;
    if (mode == 2) resid[idx] = wrap8((int)resid[idx] + val);
    else out[idx] = (int8_t)val;
  }
}

__global__ __launch_bounds__(32) void k_attn(const int8_t* __restrict__ qb,
                                             const int8_t* __restrict__ kb,
                                             const int8_t* __restrict__ vt,
                                             int8_t* __restrict__ ab) {
  extern __shared__ char smem[];
  int* scores = (int*)smem;
  char* ldsk = smem + 16 * T_ * sizeof(int);
  const int bh = blockIdx.x, b = bh >> 1, h = bh & 1;
  const int qt = blockIdx.y;
  const int lane = threadIdx.x, lo = lane & 15, hi = lane >> 4;
  const int kmax = (qt + 1) * 16;

  {
    const int8_t* kbase = kb + ((size_t)(b * T_)) * DM + h * HD;
    const int nchunk = kmax * 2;
    for (int chunk = lane; chunk < nchunk; chunk += 32) {
      int row = chunk >> 1, half = chunk & 1;
      unsigned long long gaddr =
          (unsigned long long)(uintptr_t)(kbase + (size_t)row * DM + half * 16);
      unsigned int laddr = (unsigned int)(uintptr_t)(ldsk + chunk * 16);
      asm volatile("global_load_async_to_lds_b128 %0, %1, off"
                   :: "v"(laddr), "v"(gaddr) : "memory");
    }
  }

  v8i aq;
  {
    const int8_t* p = qb + ((size_t)(b * T_ + qt * 16 + lo)) * DM + h * HD + hi * 16;
    aq[0] = *(const int*)(p + 0);  aq[1] = *(const int*)(p + 4);
    aq[2] = *(const int*)(p + 8);  aq[3] = *(const int*)(p + 12);
    aq[4] = 0; aq[5] = 0; aq[6] = 0; aq[7] = 0;
  }

  asm volatile("s_wait_asynccnt 0x0" ::: "memory");
  __syncthreads();

  for (int kt = 0; kt <= qt; ++kt) {
    v8i bk;
    const v4i kv = *(const v4i*)(ldsk + (kt * 16 + lo) * 32 + hi * 16);
    bk[0] = kv[0]; bk[1] = kv[1]; bk[2] = kv[2]; bk[3] = kv[3];
    bk[4] = 0; bk[5] = 0; bk[6] = 0; bk[7] = 0;
    v8i c = {};
    c = __builtin_amdgcn_wmma_i32_16x16x64_iu8(true, aq, true, bk, c, false, false);
#pragma unroll
    for (int v = 0; v < 8; ++v) {
      int r = v + hi * 8;
      int kidx = kt * 16 + lo;
      int s = (kidx > qt * 16 + r) ? -32767 : ((c[v] * 45) >> 8);
      scores[r * T_ + kidx] = s;
    }
  }
  __syncthreads();

  const int half = kmax >> 1;
  const int r = lo;
  const int k0 = hi * half, k1 = k0 + half;
  int m = -0x7FFFFFFF;
  for (int k = k0; k < k1; ++k) m = imax(m, scores[r * T_ + k]);
  m = imax(m, __shfl_xor(m, 16, 32));
  int ssum = 0;
  for (int k = k0; k < k1; ++k) {
    int sh = scores[r * T_ + k] - m;
    int ea = (sh >= -3) ? 256 + sh * 64
           : (sh >= -8) ? 64 + (sh + 3) * 11
           : (sh >= -24) ? sh + 24 : 0;
    ssum += ea;
    scores[r * T_ + k] = ea;
  }
  ssum += __shfl_xor(ssum, 16, 32);
  const float sdiv = (float)imax(ssum, 1);
  __syncthreads();

  const int kpad = (kmax + 63) & ~63;
  auto qp = [&](int k) -> unsigned {
    if (k >= kmax) return 0u;
    int ip = (int)rintf(((float)scores[r * T_ + k] / sdiv) * 255.f);
    return (unsigned)imin(imax(ip, 0), 255);
  };
  auto pack4 = [&](int k) -> int {
    return (int)(qp(k) | (qp(k + 1) << 8) | (qp(k + 2) << 16) | (qp(k + 3) << 24));
  };
  v8i acc0 = {}, acc1 = {};
  for (int kc = 0; kc < kpad; kc += 64) {
    v8i ap;
    const int kb8 = kc + hi * 16;
    ap[0] = pack4(kb8 + 0);  ap[1] = pack4(kb8 + 4);
    ap[2] = pack4(kb8 + 8);  ap[3] = pack4(kb8 + 12);
    ap[4] = pack4(kb8 + 32); ap[5] = pack4(kb8 + 36);
    ap[6] = pack4(kb8 + 40); ap[7] = pack4(kb8 + 44);
#pragma unroll
    for (int nt = 0; nt < 2; ++nt) {
      v8i bv;
      const int8_t* pv = vt + ((size_t)((b * NH + h) * HD + nt * 16 + lo)) * T_ + kc + hi * 16;
      bv[0] = *(const int*)(pv + 0);  bv[1] = *(const int*)(pv + 4);
      bv[2] = *(const int*)(pv + 8);  bv[3] = *(const int*)(pv + 12);
      bv[4] = *(const int*)(pv + 32); bv[5] = *(const int*)(pv + 36);
      bv[6] = *(const int*)(pv + 40); bv[7] = *(const int*)(pv + 44);
      if (nt == 0)
        acc0 = __builtin_amdgcn_wmma_i32_16x16x64_iu8(false, ap, true, bv, acc0, false, false);
      else
        acc1 = __builtin_amdgcn_wmma_i32_16x16x64_iu8(false, ap, true, bv, acc1, false, false);
    }
  }
#pragma unroll
  for (int v = 0; v < 8; ++v) {
    int rr = v + hi * 8;
    size_t row = ((size_t)(b * T_ + qt * 16 + rr)) * DM + h * HD;
    ab[row + 0 * 16 + lo] = wrap8(acc0[v] >> 8);
    ab[row + 1 * 16 + lo] = wrap8(acc1[v] >> 8);
  }
}

__global__ __launch_bounds__(32) void k_logits(const int8_t* __restrict__ X,
                                               const int8_t* __restrict__ Ehi,
                                               const int8_t* __restrict__ Elo,
                                               float* __restrict__ out) {
  const int m0 = blockIdx.x * 16, n0 = blockIdx.y * 16;
  const int lane = threadIdx.x, lo = lane & 15, hi = lane >> 4;
  v8i a, bh, bl;
  const int8_t* pa = X + (size_t)(m0 + lo) * DM + hi * 16;
  a[0] = *(const int*)(pa + 0);  a[1] = *(const int*)(pa + 4);
  a[2] = *(const int*)(pa + 8);  a[3] = *(const int*)(pa + 12);
  a[4] = *(const int*)(pa + 32); a[5] = *(const int*)(pa + 36);
  a[6] = *(const int*)(pa + 40); a[7] = *(const int*)(pa + 44);
  const size_t rowb = (size_t)(n0 + lo) * DM + hi * 16;
  const int8_t* ph = Ehi + rowb;
  const int8_t* pl = Elo + rowb;
  bh[0] = *(const int*)(ph + 0);  bh[1] = *(const int*)(ph + 4);
  bh[2] = *(const int*)(ph + 8);  bh[3] = *(const int*)(ph + 12);
  bh[4] = *(const int*)(ph + 32); bh[5] = *(const int*)(ph + 36);
  bh[6] = *(const int*)(ph + 40); bh[7] = *(const int*)(ph + 44);
  bl[0] = *(const int*)(pl + 0);  bl[1] = *(const int*)(pl + 4);
  bl[2] = *(const int*)(pl + 8);  bl[3] = *(const int*)(pl + 12);
  bl[4] = *(const int*)(pl + 32); bl[5] = *(const int*)(pl + 36);
  bl[6] = *(const int*)(pl + 40); bl[7] = *(const int*)(pl + 44);
  v8i chi = {}, clo = {};
  chi = __builtin_amdgcn_wmma_i32_16x16x64_iu8(true, a, true, bh, chi, false, false);
  clo = __builtin_amdgcn_wmma_i32_16x16x64_iu8(true, a, false, bl, clo, false, false);
#pragma unroll
  for (int v = 0; v < 8; ++v) {
    int rr = v + hi * 8;
    int acc = chi[v] * 256 + clo[v];
    out[(size_t)(m0 + rr) * VOC + n0 + lo] = (float)acc * (0.125f / 1024.f);
  }
}

extern "C" void kernel_launch(void* const* d_in, const int* in_sizes, int n_in,
                              void* d_out, int out_size, void* d_ws, size_t ws_size,
                              hipStream_t stream) {
  const int*   tokens  = (const int*)d_in[0];
  const float* tok_emb = (const float*)d_in[1];
  const float* pos_emb = (const float*)d_in[2];
  const float* attn_nw = (const float*)d_in[3];
  const float* q_w     = (const float*)d_in[4];
  const float* k_w     = (const float*)d_in[5];
  const float* v_w     = (const float*)d_in[6];
  const float* o_w     = (const float*)d_in[7];
  const float* ff_nw   = (const float*)d_in[8];
  const float* up_w    = (const float*)d_in[9];
  const float* down_w  = (const float*)d_in[10];
  const float* fin_nw  = (const float*)d_in[11];
  float* logits = (float*)d_out;

  char* base = (char*)d_ws;
  size_t off = 0;
  auto carve = [&](size_t bytes) -> char* {
    char* p = base + off;
    off = (off + bytes + 255) & ~(size_t)255;
    return p;
  };
  int8_t* wq   = (int8_t*)carve(NL * DM * DM);
  int8_t* wk   = (int8_t*)carve(NL * DM * DM);
  int8_t* wv   = (int8_t*)carve(NL * DM * DM);
  int8_t* wo   = (int8_t*)carve(NL * DM * DM);
  int8_t* wup  = (int8_t*)carve(NL * DFF * DM);
  int8_t* wdn  = (int8_t*)carve(NL * DM * DFF);
  int8_t* ehi  = (int8_t*)carve((size_t)VOC * DM);
  int8_t* elo  = (int8_t*)carve((size_t)VOC * DM);
  int8_t* x    = (int8_t*)carve((size_t)NTOK * DM);
  int8_t* hbuf = (int8_t*)carve((size_t)NTOK * DM);
  int8_t* qb   = (int8_t*)carve((size_t)NTOK * DM);
  int8_t* kb   = (int8_t*)carve((size_t)NTOK * DM);
  int8_t* vb   = (int8_t*)carve((size_t)NTOK * DM);
  int8_t* vt   = (int8_t*)carve((size_t)B_ * NH * HD * T_);
  int8_t* ab   = (int8_t*)carve((size_t)NTOK * DM);
  int8_t* ubuf = (int8_t*)carve((size_t)NTOK * DFF);
  (void)ws_size; (void)n_in; (void)in_sizes; (void)out_size;

  k_quant_ternary<<<NL, 256, 0, stream>>>(q_w, wq, DM * DM);
  k_quant_ternary<<<NL, 256, 0, stream>>>(k_w, wk, DM * DM);
  k_quant_ternary<<<NL, 256, 0, stream>>>(v_w, wv, DM * DM);
  k_quant_ternary<<<NL, 256, 0, stream>>>(o_w, wo, DM * DM);
  k_quant_ternary<<<NL, 256, 0, stream>>>(up_w, wup, DFF * DM);
  k_quant_ternary<<<NL, 256, 0, stream>>>(down_w, wdn, DM * DFF);
  k_quant_emb<<<(VOC * DM + 255) / 256, 256, 0, stream>>>(tok_emb, ehi, elo, VOC * DM);
  k_embed<<<NTOK / 256, 256, 0, stream>>>(tokens, tok_emb, pos_emb, x);

  const dim3 gProj(NTOK / 16, DM / 16);
  const dim3 gUp(NTOK / 16, DFF / 16);
  const dim3 gAttn(B_ * NH, T_ / 16);
  const size_t attnLds = 16 * T_ * sizeof(int) + (size_t)T_ * 32;

  for (int i = 0; i < NL; ++i) {
    k_rmsnorm<<<NTOK / 256, 256, 0, stream>>>(x, attn_nw + i * DM, hbuf);
    k_bitlinear<<<gProj, 32, 0, stream>>>(hbuf, wq + i * DM * DM, qb, nullptr, DM, 0);
    k_bitlinear<<<gProj, 32, 0, stream>>>(hbuf, wk + i * DM * DM, kb, nullptr, DM, 0);
    k_bitlinear<<<gProj, 32, 0, stream>>>(hbuf, wv + i * DM * DM, vb, nullptr, DM, 0);
    k_vtrans<<<NTOK / 256, 256, 0, stream>>>(vb, vt);
    k_attn<<<gAttn, 32, attnLds, stream>>>(qb, kb, vt, ab);
    k_bitlinear<<<gProj, 32, 0, stream>>>(ab, wo + i * DM * DM, nullptr, x, DM, 2);
    k_rmsnorm<<<NTOK / 256, 256, 0, stream>>>(x, ff_nw + i * DM, hbuf);
    k_bitlinear<<<gUp, 32, 0, stream>>>(hbuf, wup + i * DFF * DM, ubuf, nullptr, DM, 1);
    k_bitlinear<<<gProj, 32, 0, stream>>>(ubuf, wdn + i * DM * DFF, nullptr, x, DFF, 2);
  }
  k_rmsnorm<<<NTOK / 256, 256, 0, stream>>>(x, fin_nw, hbuf);
  k_logits<<<dim3(NTOK / 16, VOC / 16), 32, 0, stream>>>(hbuf, ehi, elo, logits);
}
